// SAGEFC_75849122447577
// MI455X (gfx1250) — compile-verified, matrix operand order checked by reading, not run on hardware
//
#include <hip/hip_runtime.h>
#include <stdint.h>

// ---------------------------------------------------------------------------
// GraphSAGE (3-layer) for MI455X / gfx1250, wave32.
//  - mean-aggregation: fp32 global atomics (memory-bound phase, ~8.4 GB/pass)
//  - dense GEMMs: v_wmma_f32_16x16x32_bf16; block = 4 waves sharing one staged
//    A tile (64 rows x 64 cols per block) -> A operand traffic / 4
//  - weights pre-packed fragment-major: B fragment = 2x global_load_b128
//  - cheap bf16 pack (round-to-nearest, ties-away) -> no 16-bit insert chains
//  - all 8 fragment ds_loads issued before the 4 back-to-back WMMAs
// ---------------------------------------------------------------------------

typedef __attribute__((ext_vector_type(16))) __bf16   v16bf;
typedef __attribute__((ext_vector_type(2)))  __bf16   v2bf;
typedef __attribute__((ext_vector_type(8)))  float    v8f;
typedef __attribute__((ext_vector_type(8)))  uint32_t v8u;

#define GNN_IN  128
#define GNN_HID 256
#define GNN_OUT 47
#define MT 4                 // 16-row tiles per wave (64 rows per block)
#define NW 4                 // waves per block (4 column tiles share A stage)
#define LDSW 20              // LDS row stride in dwords (16 + 4 pad, 16B align)

__device__ __forceinline__ uint32_t pack_bf16(float lo, float hi) {
#if __has_builtin(__builtin_amdgcn_cvt_pk_bf16_f32)
  v2bf r = __builtin_amdgcn_cvt_pk_bf16_f32(lo, hi);
  return __builtin_bit_cast(uint32_t, r);
#else
  // round-to-nearest (ties away from zero): 2 independent adds + 1 combine
  uint32_t a = __float_as_uint(lo) + 0x8000u;
  uint32_t b = __float_as_uint(hi) + 0x8000u;
#if __has_builtin(__builtin_amdgcn_perm)
  return __builtin_amdgcn_perm(b, a, 0x07060302u);  // {b[31:16], a[31:16]}
#else
  return (a >> 16) | (b & 0xFFFF0000u);
#endif
#endif
}

// ------------------------------ utility kernels ----------------------------

__global__ void __launch_bounds__(256) zero_kernel(float* __restrict__ p, long long n) {
  long long i = (long long)blockIdx.x * blockDim.x + threadIdx.x;
  if (i < n) p[i] = 0.0f;
}

__global__ void __launch_bounds__(256) degree_kernel(const int* __restrict__ dst,
                                                     float* __restrict__ deg, int E) {
  int e = blockIdx.x * blockDim.x + threadIdx.x;
  if (e < E) atomicAdd(&deg[dst[e]], 1.0f);
}

__global__ void __launch_bounds__(256) invdeg_kernel(float* __restrict__ deg, int N) {
  int i = blockIdx.x * blockDim.x + threadIdx.x;
  if (i < N) deg[i] = 1.0f / fmaxf(deg[i], 1.0f);
}

// scatter-add of source features into destination accumulator (fp32 atomics)
__global__ void __launch_bounds__(256) scatter_add_kernel(
    const float* __restrict__ feat, const int* __restrict__ src,
    const int* __restrict__ dst, float* __restrict__ out,
    int K, int k4shift, long long total, int relu) {
  long long idx = (long long)blockIdx.x * blockDim.x + threadIdx.x;
  if (idx >= total) return;
  int e = (int)(idx >> k4shift);
  int q = (int)(idx & ((1 << k4shift) - 1));
  int s = src[e], d = dst[e];
  float4 v = *(const float4*)(feat + (size_t)s * K + (size_t)q * 4);
  if (relu) {
    v.x = fmaxf(v.x, 0.f); v.y = fmaxf(v.y, 0.f);
    v.z = fmaxf(v.z, 0.f); v.w = fmaxf(v.w, 0.f);
  }
  float* o = out + (size_t)d * K + (size_t)q * 4;
  atomicAdd(o + 0, v.x); atomicAdd(o + 1, v.y);
  atomicAdd(o + 2, v.z); atomicAdd(o + 3, v.w);
}

// Pack fp32 weight [K,O] into fragment-major bf16 layout:
//   Wp[((kt*ctiles + ct)*32 + lane)*8 + j]
// holds {W[2*kpg][c], W[2*kpg+1][c]} where
//   kpbase = (lane<16) ? 0 : 4
//   kpg    = kt*16 + kpbase + (j<4 ? j : 4 + j)   // {0..3, 8..11} pattern
//   c      = ct*16 + (lane&15)                    // zero-padded past O
__global__ void __launch_bounds__(256) pack_weights_kernel(
    const float* __restrict__ W, uint32_t* __restrict__ Wp,
    int ktiles, int ctiles, int O) {
  int idx = blockIdx.x * blockDim.x + threadIdx.x;
  int total = ktiles * ctiles * 256;
  if (idx >= total) return;
  int j    = idx & 7;
  int lane = (idx >> 3) & 31;
  int ct   = (idx >> 8) % ctiles;
  int kt   = idx / (ctiles * 256);
  int kpbase = (lane < 16) ? 0 : 4;
  int kpg = kt * 16 + kpbase + ((j < 4) ? j : 4 + j);
  int c = ct * 16 + (lane & 15);
  float lo = (c < O) ? W[(size_t)(2 * kpg) * O + c] : 0.0f;
  float hi = (c < O) ? W[(size_t)(2 * kpg + 1) * O + c] : 0.0f;
  Wp[idx] = pack_bf16(lo, hi);
}

// ------------------------------ fused SAGE GEMM ----------------------------
// One block = 4 waves; each wave owns one 16-col tile, all share the staged
// 64x32 A tile in LDS. Each wave computes a 64x16 output tile (4 WMMAs/K-step):
//   Out = act( (Asum*invdeg) @ Wl + bias + act_x(X) @ Wr )
// Fragment mappings (ISA 7.12.2, 16-bit A 16x32 / f32 C 16x16):
//   A: lane holds row m=lane&15; kp = {kpbase..+3, kpbase+8..+11}, kpbase=(lane<16)?0:4
//   B: mirrored, column c = lane&15 (pre-packed fragment-major in memory)
//   C/D: VGPR j -> row (lane<16 ? j : 8+j), col = lane&15

__device__ __forceinline__ void sage_gemm_operand(
    v8f (&acc)[MT], const float* __restrict__ src, int K,
    const float* __restrict__ invdeg, int mode,  // 0: *invdeg, 1: identity, 2: relu
    const uint32_t* __restrict__ Wfrag, int ctiles, int cblk, bool valid,
    int tile_m, int tid, int N, uint32_t (&lds)[64][LDSW]) {
  const int lane = tid & 31;
  const int m = lane & 15;
  const int kpbase = (lane < 16) ? 0 : 4;
  const int srow = tid >> 1;                 // staged row 0..63 (2 threads/row)
  const int half = tid & 1;                  // which 16-float half of the row
  const int gr = min(tile_m + srow, N - 1);
  const float s = (mode == 0) ? invdeg[gr] : 1.0f;
  const float* __restrict__ rp = src + (size_t)gr * K + half * 16;
  const int ktiles = K >> 5;

  for (int kt = 0; kt < ktiles; ++kt) {
    const int kb = kt << 5;
    // ---- stage: 16 floats per thread -> 8 bf16 pairs in LDS ----
    float4 f[4];
#pragma unroll
    for (int j = 0; j < 4; ++j) f[j] = *(const float4*)(rp + kb + 4 * j);
#pragma unroll
    for (int j = 0; j < 4; ++j) {
      float4 a = f[j];
      a.x *= s; a.y *= s; a.z *= s; a.w *= s;
      if (mode == 2) {
        a.x = fmaxf(a.x, 0.f); a.y = fmaxf(a.y, 0.f);
        a.z = fmaxf(a.z, 0.f); a.w = fmaxf(a.w, 0.f);
      }
      f[j] = a;
    }
    uint4 p0, p1;
    p0.x = pack_bf16(f[0].x, f[0].y); p0.y = pack_bf16(f[0].z, f[0].w);
    p0.z = pack_bf16(f[1].x, f[1].y); p0.w = pack_bf16(f[1].z, f[1].w);
    p1.x = pack_bf16(f[2].x, f[2].y); p1.y = pack_bf16(f[2].z, f[2].w);
    p1.z = pack_bf16(f[3].x, f[3].y); p1.w = pack_bf16(f[3].z, f[3].w);
    *(uint4*)&lds[srow][half * 8 + 0] = p0;
    *(uint4*)&lds[srow][half * 8 + 4] = p1;
    __syncthreads();

    if (valid) {  // wave-uniform: EXEC stays all-ones around WMMA
      // ---- B fragment: two coalesced uint4 loads (fragment-major pack) ----
      const uint32_t* wp = Wfrag + (((size_t)kt * ctiles + cblk) * 32 + lane) * 8;
      uint4 bq0 = *(const uint4*)(wp);
      uint4 bq1 = *(const uint4*)(wp + 4);
      v8u bu = {bq0.x, bq0.y, bq0.z, bq0.w, bq1.x, bq1.y, bq1.z, bq1.w};
      if (kt + 1 < ktiles)  // hint next K-tile of weights (global_prefetch_b8)
        __builtin_prefetch(wp + (size_t)ctiles * 256, 0, 1);

      // ---- issue all 8 fragment ds_loads, then 4 back-to-back WMMAs ----
      v8u au[MT];
#pragma unroll
      for (int t = 0; t < MT; ++t) {
        const int row = t * 16 + m;
        uint4 aq0 = *(const uint4*)&lds[row][kpbase];      // kp kpbase..+3
        uint4 aq1 = *(const uint4*)&lds[row][kpbase + 8];  // kp kpbase+8..+11
        au[t] = v8u{aq0.x, aq0.y, aq0.z, aq0.w, aq1.x, aq1.y, aq1.z, aq1.w};
      }
#pragma unroll
      for (int t = 0; t < MT; ++t) {
        acc[t] = __builtin_amdgcn_wmma_f32_16x16x32_bf16(
            false, __builtin_bit_cast(v16bf, au[t]),
            false, __builtin_bit_cast(v16bf, bu),
            (short)0, acc[t], false, false);
      }
    }
    __syncthreads();
  }
}

__global__ void __launch_bounds__(NW * 32) sage_gemm_kernel(
    const float* __restrict__ Asum, const float* __restrict__ invdeg,
    const float* __restrict__ X,
    const uint32_t* __restrict__ WlP, const uint32_t* __restrict__ WrP,
    const float* __restrict__ bias, float* __restrict__ Out,
    int KA, int KX, int O, int ctiles, int N, int relu_out, int relu_x) {
  __shared__ uint32_t lds[64][LDSW];
  const int tid = threadIdx.x;
  const int wave = tid >> 5;
  const int lane = tid & 31;
  const int tile_m = blockIdx.x * 64;
  const int cblk = blockIdx.y * NW + wave;
  const bool valid = cblk < ctiles;
  const int c = cblk * 16 + (lane & 15);

  v8f acc[MT] = {};
  sage_gemm_operand(acc, Asum, KA, invdeg, 0, WlP, ctiles, cblk, valid,
                    tile_m, tid, N, lds);
  sage_gemm_operand(acc, X, KX, invdeg, relu_x ? 2 : 1, WrP, ctiles, cblk, valid,
                    tile_m, tid, N, lds);

  if (valid) {
    const float b = (c < O) ? bias[c] : 0.0f;
#pragma unroll
    for (int t = 0; t < MT; ++t) {
      const int rbase = tile_m + t * 16 + ((lane < 16) ? 0 : 8);
#pragma unroll
      for (int j = 0; j < 8; ++j) {
        float v = acc[t][j] + b;
        if (relu_out) v = fmaxf(v, 0.0f);
        int row = rbase + j;
        if (row < N && c < O) Out[(size_t)row * O + c] = v;
      }
    }
  }
}

// --------------------------------- launcher --------------------------------

extern "C" void kernel_launch(void* const* d_in, const int* in_sizes, int n_in,
                              void* d_out, int out_size, void* d_ws, size_t ws_size,
                              hipStream_t stream) {
  const int N = in_sizes[0] / GNN_IN;   // 100000
  const int E = in_sizes[1] / 2;        // 1600000

  const float* x   = (const float*)d_in[0];
  const int*   ei  = (const int*)d_in[1];   // int32 (JAX canonicalizes int64->int32)
  const int*   src = ei;
  const int*   dst = ei + E;
  const float* Wl0 = (const float*)d_in[2];
  const float* bl0 = (const float*)d_in[3];
  const float* Wr0 = (const float*)d_in[4];
  const float* Wl1 = (const float*)d_in[5];
  const float* bl1 = (const float*)d_in[6];
  const float* Wr1 = (const float*)d_in[7];
  const float* Wl2 = (const float*)d_in[8];
  const float* bl2 = (const float*)d_in[9];
  const float* Wr2 = (const float*)d_in[10];

  float* outp = (float*)d_out;                       // [N,47]
  float* hout = outp + (size_t)N * GNN_OUT;          // [N,256] pre-ReLU hidden

  // workspace layout (~206 MB): deg | bufA(N*256) | bufB(N*256) | packed weights
  float* deg  = (float*)d_ws;
  float* bufA = deg + N;
  float* bufB = bufA + (size_t)N * GNN_HID;
  uint32_t* WlP0 = (uint32_t*)(bufB + (size_t)N * GNN_HID);  // 4*16*256
  uint32_t* WrP0 = WlP0 + 4 * 16 * 256;
  uint32_t* WlP1 = WrP0 + 4 * 16 * 256;                      // 8*16*256
  uint32_t* WrP1 = WlP1 + 8 * 16 * 256;
  uint32_t* WlP2 = WrP1 + 8 * 16 * 256;                      // 8*3*256
  uint32_t* WrP2 = WlP2 + 8 * 3 * 256;

  const int CT_H = GNN_HID / 16;        // 16 column tiles for O=256
  const int CT_O = 3;                   // 47 -> 3 column tiles
  const int Mblocks = (N + 63) / 64;
  const int YB_H = (CT_H + NW - 1) / NW;
  const int YB_O = (CT_O + NW - 1) / NW;

  auto blocks = [](long long n, int t) { return (unsigned)((n + t - 1) / t); };

  // 0) zero deg + bufA (atomic accumulators)
  {
    long long n = (long long)N + (long long)N * GNN_HID;
    zero_kernel<<<blocks(n, 256), 256, 0, stream>>>(deg, n);
  }
  // 1) pack all six weight matrices into fragment-major WMMA B layout
  pack_weights_kernel<<<blocks(4 * 16 * 256, 256), 256, 0, stream>>>(Wl0, WlP0, 4, 16, GNN_HID);
  pack_weights_kernel<<<blocks(4 * 16 * 256, 256), 256, 0, stream>>>(Wr0, WrP0, 4, 16, GNN_HID);
  pack_weights_kernel<<<blocks(8 * 16 * 256, 256), 256, 0, stream>>>(Wl1, WlP1, 8, 16, GNN_HID);
  pack_weights_kernel<<<blocks(8 * 16 * 256, 256), 256, 0, stream>>>(Wr1, WrP1, 8, 16, GNN_HID);
  pack_weights_kernel<<<blocks(8 * 3 * 256, 256), 256, 0, stream>>>(Wl2, WlP2, 8, CT_O, GNN_OUT);
  pack_weights_kernel<<<blocks(8 * 3 * 256, 256), 256, 0, stream>>>(Wr2, WrP2, 8, CT_O, GNN_OUT);
  // 2) degree -> invdeg
  degree_kernel<<<blocks(E, 256), 256, 0, stream>>>(dst, deg, E);
  invdeg_kernel<<<blocks(N, 256), 256, 0, stream>>>(deg, N);

  // ---- layer 0: h1 = relu(mean(x) @ Wl0 + bl0 + x @ Wr0) ----
  {
    long long total = (long long)E * (GNN_IN / 4);
    scatter_add_kernel<<<blocks(total, 256), 256, 0, stream>>>(
        x, src, dst, bufA, GNN_IN, 5, total, 0);
    dim3 g(Mblocks, YB_H);
    sage_gemm_kernel<<<g, NW * 32, 0, stream>>>(bufA, deg, x, WlP0, WrP0, bl0, bufB,
                                                GNN_IN, GNN_IN, GNN_HID, CT_H, N,
                                                /*relu_out=*/1, /*relu_x=*/0);
  }
  // ---- layer 1: h_out = mean(h1) @ Wl1 + bl1 + h1 @ Wr1 (no relu; -> d_out) ----
  {
    long long nz = (long long)N * GNN_HID;
    zero_kernel<<<blocks(nz, 256), 256, 0, stream>>>(bufA, nz);
    long long total = (long long)E * (GNN_HID / 4);
    scatter_add_kernel<<<blocks(total, 256), 256, 0, stream>>>(
        bufB, src, dst, bufA, GNN_HID, 6, total, 0);
    dim3 g(Mblocks, YB_H);
    sage_gemm_kernel<<<g, NW * 32, 0, stream>>>(bufA, deg, bufB, WlP1, WrP1, bl1, hout,
                                                GNN_HID, GNN_HID, GNN_HID, CT_H, N,
                                                /*relu_out=*/0, /*relu_x=*/0);
  }
  // ---- layer 2: out = mean(relu(h_out)) @ Wl2 + bl2 + relu(h_out) @ Wr2 ----
  {
    long long nz = (long long)N * GNN_HID;
    zero_kernel<<<blocks(nz, 256), 256, 0, stream>>>(bufB, nz);
    long long total = (long long)E * (GNN_HID / 4);
    scatter_add_kernel<<<blocks(total, 256), 256, 0, stream>>>(
        hout, src, dst, bufB, GNN_HID, 6, total, /*relu=*/1);
    dim3 g(Mblocks, YB_O);
    sage_gemm_kernel<<<g, NW * 32, 0, stream>>>(bufB, deg, hout, WlP2, WrP2, bl2, outp,
                                                GNN_HID, GNN_HID, GNN_OUT, CT_O, N,
                                                /*relu_out=*/0, /*relu_x=*/1);
  }
}
